// LSTM_24111946400314
// MI455X (gfx1250) — matrix operand order repaired and checked by reading, not compiled, not run on hardware
//
#include <hip/hip_runtime.h>
#include <hip/hip_bf16.h>

// ---------------------------------------------------------------------------
// LSTM  (T=65536, IN=128, H=100, OUT=1, batch=1)  for MI455X / gfx1250
//
// Phase 0: one-time f32 -> f16 conversion of x and W_ih (halves GEMM traffic)
// Phase 1: x_proj[T,400] = x[T,128] @ W_ih^T + bias  -- WMMA f16->f32,
//          tiles staged into LDS via the Tensor Data Mover (tensor_load_to_lds)
// Phase 2: sequential LSTM scan, one persistent workgroup (latency-bound part)
// Phase 3: pred[T] = sigmoid(hs @ W_lin^T + b_lin)
// ---------------------------------------------------------------------------

typedef __attribute__((ext_vector_type(16))) _Float16     v16h;
typedef __attribute__((ext_vector_type(8)))  float        v8f;
typedef __attribute__((ext_vector_type(4)))  unsigned int v4u;
typedef __attribute__((ext_vector_type(8)))  int          v8i_;
typedef __attribute__((ext_vector_type(4)))  int          v4i_;

#define H4   400   // 4*HIDDEN
#define HID  100
#define INSZ 128
#define JT   80    // gate-columns per block in phase 1 (5 wmma N-tiles)

#if defined(__has_builtin)
#if __has_builtin(__builtin_amdgcn_tensor_load_to_lds) && \
    __has_builtin(__builtin_amdgcn_s_wait_tensorcnt)
#define USE_TDM 1
#endif
#endif

// ---------------------------------------------------------------------------
// Phase 0: elementwise f32 -> f16
// ---------------------------------------------------------------------------
__global__ __launch_bounds__(256) void cvt_f16(
    const float* __restrict__ s, _Float16* __restrict__ d, int n4)
{
    int i = blockIdx.x * 256 + threadIdx.x;
    if (i < n4) {
        float4 v = ((const float4*)s)[i];
        d[4 * i + 0] = (_Float16)v.x;
        d[4 * i + 1] = (_Float16)v.y;
        d[4 * i + 2] = (_Float16)v.z;
        d[4 * i + 3] = (_Float16)v.w;
    }
}

#ifdef USE_TDM
// Build D# (group0 128b + group1 256b) for a 2-D f16 tile copy and issue the
// TDM DMA.  tensor dims == tile dims (tile fully in-bounds), dim0 stride = row
// width, data_size=1 (2 bytes).  Groups 2/3 zero (<=2D tensor).
// This toolchain exposes the 6-arg builtin:
//   (u32x4 g0, i32x8 g1, i32x4 g2, i32x4 g3, i32x8 extra, i32 cpol)
__device__ __forceinline__ void tdm_load_2d_f16(
    unsigned lds_addr, const void* gptr, unsigned width, unsigned rows)
{
    unsigned long long ga = (unsigned long long)(uintptr_t)gptr;
    v4u g0;
    g0[0] = 1u;                                                // count=1, user
    g0[1] = lds_addr;                                          // lds_addr
    g0[2] = (unsigned)(ga & 0xFFFFFFFFu);                      // global_addr lo
    g0[3] = (unsigned)((ga >> 32) & 0x01FFFFFFu) | (2u << 30); // addr hi | type=2
    v8i_ g1;
    g1[0] = (int)(1u << 16);                                   // data_size = 2B
    g1[1] = (int)((width & 0xFFFFu) << 16);                    // tensor_dim0 lo
    g1[2] = (int)(((width >> 16) & 0xFFFFu) |
                  ((rows & 0xFFFFu) << 16));                   // dim0 hi | dim1 lo
    g1[3] = (int)(((rows >> 16) & 0xFFFFu) |
                  ((width & 0xFFFFu) << 16));                  // dim1 hi | tile_dim0
    g1[4] = (int)(rows & 0xFFFFu);                             // tile_dim1, tile_dim2=0
    g1[5] = (int)width;                                        // dim0_stride lo32
    g1[6] = 0;                                                 // stride hi / dim1_stride
    g1[7] = 0;
    v4i_ z4 = {0, 0, 0, 0};
    v8i_ z8 = {0, 0, 0, 0, 0, 0, 0, 0};
    __builtin_amdgcn_tensor_load_to_lds(g0, g1, z4, z4, z8, 0);
}
#endif

// ---------------------------------------------------------------------------
// Phase 1: tiled WMMA GEMM.  Block = 256 threads (8 waves).
// Block tile: 128 timesteps x 80 gate-columns; K=128 (4 wmma k-steps).
// Each wave: 16 t-rows, reuses one A fragment across 5 N-tiles per k-step
// -> 20 v_wmma per wave.  A(128x128 f16)+B(80x128 f16) staged via TDM.
// ---------------------------------------------------------------------------
__global__ __launch_bounds__(256) void xproj_wmma(
    const _Float16* __restrict__ xh, const _Float16* __restrict__ wh,
    const float* __restrict__ bih, const float* __restrict__ bhh,
    float* __restrict__ xp, int T)
{
    __shared__ _Float16 Ah[128 * 128];   // [t][k]  32 KB
    __shared__ _Float16 Bh[JT * 128];    // [j][k]  20 KB

    const int tid = threadIdx.x;
    const int t0  = blockIdx.x * 128;
    const int j0  = blockIdx.y * JT;

#ifdef USE_TDM
    if (tid < 32) {                      // one wave issues the DMAs (EXEC ignored)
        tdm_load_2d_f16((unsigned)(uintptr_t)(void*)Ah,
                        xh + (size_t)t0 * INSZ, INSZ, 128);
        tdm_load_2d_f16((unsigned)(uintptr_t)(void*)Bh,
                        wh + (size_t)j0 * INSZ, INSZ, JT);
        __builtin_amdgcn_s_wait_tensorcnt(0);
    }
    __syncthreads();
#else
    {   // fallback: coalesced uint4 staging
        const uint4* ag = (const uint4*)(xh + (size_t)t0 * INSZ);
        uint4* As = (uint4*)Ah;
#pragma unroll
        for (int i = 0; i < 8; ++i) As[i * 256 + tid] = ag[i * 256 + tid];
        const uint4* bg = (const uint4*)(wh + (size_t)j0 * INSZ);
        uint4* Bs = (uint4*)Bh;
#pragma unroll
        for (int i = 0; i < 5; ++i) Bs[i * 256 + tid] = bg[i * 256 + tid];
        __syncthreads();
    }
#endif

    const int lane = tid & 31;
    const int wv   = tid >> 5;     // wave id 0..7 -> 16 t-rows each
    const int m    = lane & 15;
    const int hi   = lane >> 4;

    union Frag { v16h h; unsigned int u[8]; };

    v8f acc[5] = {};
#pragma unroll
    for (int ks = 0; ks < 4; ++ks) {
        const int k0 = ks * 32;
        Frag a;
#pragma unroll
        for (int v = 0; v < 8; ++v) {
            // A 16x32 f16 layout: v<4 -> K=2v+8*hi ; v>=4 -> K=16+2(v-4)+8*hi
            int ka = (v < 4) ? (2 * v + 8 * hi) : (16 + 2 * (v - 4) + 8 * hi);
            a.u[v] = *(const unsigned int*)&Ah[(wv * 16 + m) * 128 + k0 + ka];
        }
#pragma unroll
        for (int n = 0; n < 5; ++n) {
            Frag b;
#pragma unroll
            for (int v = 0; v < 8; ++v) {
                // B 32x16 f16 layout: v<4 -> K=2v+8*hi ; v>=4 -> K=16+2(v-4)+8*hi, N = lane%16
                int kb = (v < 4) ? (2 * v + 8 * hi) : (16 + 2 * (v - 4) + 8 * hi);
                b.u[v] = *(const unsigned int*)&Bh[(n * 16 + m) * 128 + k0 + kb];
            }
            acc[n] = __builtin_amdgcn_wmma_f32_16x16x32_f16(
                false, a.h, false, b.h, (short)0, acc[n], false, false);
        }
    }

#pragma unroll
    for (int n = 0; n < 5; ++n) {
        const int j = j0 + n * 16 + m;
        const float bias = bih[j] + bhh[j];
        // C layout: VGPR r -> M = r + 8*hi, N = lane%16
#pragma unroll
        for (int r = 0; r < 8; ++r) {
            int row = t0 + wv * 16 + r + 8 * hi;
            xp[(size_t)row * H4 + j] = acc[n][r] + bias;
        }
    }
}

// ---------------------------------------------------------------------------
// Phase 2: persistent single-workgroup scan.  512 threads (16 wave32).
//   thread r<400 : gates[r] = xp[t][r] + dot(h, W_hh[r][:])   (W row in VGPRs)
//   thread r<100 : gate combine -> c (register), h -> LDS + global hs
// ---------------------------------------------------------------------------
__device__ __forceinline__ float fast_sigmoid(float x) {
    return 1.0f / (1.0f + __expf(-x));
}
__device__ __forceinline__ float fast_tanh(float x) {
    float e = __expf(2.0f * x);
    return (e - 1.0f) / (e + 1.0f);
}

__global__ __launch_bounds__(512, 1) void lstm_scan(
    const float* __restrict__ xp, const float* __restrict__ Whh,
    float* __restrict__ hs, int T)
{
    __shared__ __align__(16) float h_lds[104];
    __shared__ float gbuf[H4];

    const int r = threadIdx.x;

    float4 w4[25];                       // W_hh row resident in VGPRs
    if (r < H4) {
#pragma unroll
        for (int q = 0; q < 25; ++q)
            w4[q] = *(const float4*)&Whh[r * HID + 4 * q];
    }
    if (r < 104) h_lds[r] = 0.0f;
    float c = 0.0f;
    float xp_cur = (r < H4) ? xp[r] : 0.0f;
    __syncthreads();

    for (int t = 0; t < T; ++t) {
        float xp_next = 0.0f;
        if (r < H4 && (t + 1) < T) xp_next = xp[(size_t)(t + 1) * H4 + r];
        if (((t & 7) == 0) && r < H4 && ((r & 15) == 0) && (t + 8) < T)
            __builtin_prefetch(&xp[(size_t)(t + 8) * H4 + r], 0, 0);

        if (r < H4) {
            float s0 = xp_cur, s1 = 0.0f, s2 = 0.0f, s3 = 0.0f;
#pragma unroll
            for (int q = 0; q < 25; ++q) {
                float4 h4 = *(const float4*)&h_lds[4 * q];
                s0 = __builtin_fmaf(h4.x, w4[q].x, s0);
                s1 = __builtin_fmaf(h4.y, w4[q].y, s1);
                s2 = __builtin_fmaf(h4.z, w4[q].z, s2);
                s3 = __builtin_fmaf(h4.w, w4[q].w, s3);
            }
            gbuf[r] = (s0 + s1) + (s2 + s3);
        }
        __syncthreads();

        if (r < HID) {
            float ig = fast_sigmoid(gbuf[r]);
            float fg = fast_sigmoid(gbuf[r + 100]);
            float gg = fast_tanh(gbuf[r + 200]);
            float og = fast_sigmoid(gbuf[r + 300]);
            c = fg * c + ig * gg;
            float hn = og * fast_tanh(c);
            h_lds[r] = hn;
            hs[(size_t)t * HID + r] = hn;
        }
        __syncthreads();
        xp_cur = xp_next;
    }
}

// ---------------------------------------------------------------------------
// Phase 3: pred[t] = sigmoid(dot(hs[t], W_lin) + b_lin)
// ---------------------------------------------------------------------------
__global__ __launch_bounds__(256) void lstm_head(
    const float* __restrict__ hs, const float* __restrict__ Wlin,
    const float* __restrict__ blin, float* __restrict__ out)
{
    __shared__ float ht[32 * HID];
    __shared__ float wl[HID];
    const int tid = threadIdx.x;
    const size_t base = (size_t)blockIdx.x * (32 * HID);

    for (int i = tid; i < 32 * HID; i += 256) ht[i] = hs[base + i];
    if (tid < HID) wl[tid] = Wlin[tid];
    __syncthreads();

    if (tid < 32) {
        float acc = blin[0];
#pragma unroll
        for (int k = 0; k < HID; ++k)
            acc = __builtin_fmaf(ht[tid * HID + k], wl[k], acc);
        out[blockIdx.x * 32 + tid] = 1.0f / (1.0f + __expf(-acc));
    }
}

// ---------------------------------------------------------------------------
extern "C" void kernel_launch(void* const* d_in, const int* in_sizes, int n_in,
                              void* d_out, int out_size, void* d_ws, size_t ws_size,
                              hipStream_t stream)
{
    const float* x    = (const float*)d_in[0];   // [T,128]
    const float* Wih  = (const float*)d_in[1];   // [400,128]
    const float* Whh  = (const float*)d_in[2];   // [400,100]
    const float* bih  = (const float*)d_in[3];   // [400]
    const float* bhh  = (const float*)d_in[4];   // [400]
    const float* Wlin = (const float*)d_in[5];   // [100]
    const float* blin = (const float*)d_in[6];   // [1]
    float* out = (float*)d_out;                  // [T]

    const int T = in_sizes[0] / INSZ;            // 65536

    // workspace layout
    float*    xp    = (float*)d_ws;                       // [T,400] f32
    float*    hs    = xp + (size_t)T * H4;                // [T,100] f32
    _Float16* xh    = (_Float16*)(hs + (size_t)T * HID);  // [T,128] f16
    _Float16* wih_h = xh + (size_t)T * INSZ;              // [400,128] f16

    // Phase 0: f32 -> f16
    int n4x = T * INSZ / 4;
    cvt_f16<<<(n4x + 255) / 256, 256, 0, stream>>>(x, xh, n4x);
    int n4w = H4 * INSZ / 4;
    cvt_f16<<<(n4w + 255) / 256, 256, 0, stream>>>(Wih, wih_h, n4w);

    // Phase 1: WMMA GEMM
    dim3 g1(T / 128, H4 / JT);                   // 512 x 5
    xproj_wmma<<<g1, 256, 0, stream>>>(xh, wih_h, bih, bhh, xp, T);

    // Phase 2: sequential scan
    lstm_scan<<<1, 512, 0, stream>>>(xp, Whh, hs, T);

    // Phase 3: output head
    lstm_head<<<T / 32, 256, 0, stream>>>(hs, Wlin, blin, out);
}
